// VectorQuantizer_38062000177876
// MI455X (gfx1250) — matrix operand order repaired and checked by reading, not compiled, not run on hardware
//
#include <hip/hip_runtime.h>

typedef __attribute__((ext_vector_type(16))) _Float16 v16h;
typedef __attribute__((ext_vector_type(8)))  float    v8f;

#define K_CODES 2048
#define C_DIM   256
#define N_VECS  65536   // 64 * 32 * 32
#define HW      1024    // 32*32
#define CHW     262144  // 256*1024
#define TILE_CODES 32   // codes staged per LDS buffer

// async global->LDS copy of 16 bytes per lane (gfx1250, ASYNCcnt-tracked).
// GVS addressing: 64-bit SGPR base + 32-bit per-lane unsigned byte offset.
// The 24-bit instruction offset is added to BOTH the LDS and global addresses.
#define ASYNC_CP16(ldsaddr, goff, base, OFF)                                   \
    asm volatile("global_load_async_to_lds_b128 %0, %1, %2 offset:" #OFF      \
                 :: "v"(ldsaddr), "v"(goff), "s"(base) : "memory")

__device__ __forceinline__ void wait_async0() {
#if __has_builtin(__builtin_amdgcn_s_wait_asynccnt)
    __builtin_amdgcn_s_wait_asynccnt(0);
#else
    asm volatile("s_wait_asynccnt 0x0" ::: "memory");
#endif
}

// ---------------------------------------------------------------------------
// Kernel 1: codebook -> f16 copy + per-code squared L2 norm; zero accumulators
// ---------------------------------------------------------------------------
__global__ void vq_prep_kernel(const float* __restrict__ codebook,
                               _Float16* __restrict__ cb16,
                               float* __restrict__ cnorm,
                               int* __restrict__ counts,
                               float* __restrict__ lossacc) {
    const int k = blockIdx.x;
    const int t = threadIdx.x;
    const float v = codebook[k * C_DIM + t];
    cb16[k * C_DIM + t] = (_Float16)v;

    __shared__ float red[256];
    red[t] = v * v;
    __syncthreads();
    for (int s = 128; s > 0; s >>= 1) {
        if (t < s) red[t] += red[t + s];
        __syncthreads();
    }
    if (t == 0) {
        cnorm[k]  = red[0];
        counts[k] = 0;
        if (k == 0) *lossacc = 0.0f;
    }
}

// ---------------------------------------------------------------------------
// Kernel 2: WMMA distance GEMM + fused argmin, async-LDS staged B.
// Block = 128 threads (4 waves); each wave owns 16 rows of Z, all waves share
// the same code sweep, so B tiles are staged ONCE per block into LDS via
// global_load_async_to_lds_b128 (double-buffered, overlapped with compute).
//
// A (16x32 f16): lane l<16 -> row M=l, K = k0+{0..15};
//                lane l>=16 -> row M=l-16, K = k0+{16..31}.
// B (32x16 f16): lane l -> column N=l%16, K = k0 + (l/16)*16 + {0..15}
// C/D (16x16 f32): VGPR r, lane l -> M = r + 8*(l/16), N = l%16.
// ---------------------------------------------------------------------------
__global__ void vq_argmin_kernel(const float* __restrict__ z,
                                 const _Float16* __restrict__ cb16,
                                 const float* __restrict__ cnorm,
                                 int* __restrict__ idx_out) {
    __shared__ alignas(32) _Float16 Bbuf[2][TILE_CODES * C_DIM];  // 2 x 16 KB

    const int tid  = threadIdx.x;
    const int lane = tid & 31;
    const int wave = tid >> 5;
    const int tile = blockIdx.x * (blockDim.x >> 5) + wave;
    const int rowBase = tile * 16;
    const int half = lane >> 4;   // 0 or 1
    const int sub  = lane & 15;

    // ---- Load + convert this wave's 16 rows of Z into 8 A-fragments (once) --
    const int n = rowBase + sub;
    const float* __restrict__ zrow = z + (size_t)(n >> 10) * CHW + (n & (HW - 1));
    v16h A[8];
#pragma unroll
    for (int kc = 0; kc < 8; ++kc) {
        const int k0 = kc * 32;
#pragma unroll
        for (int e = 0; e < 8; ++e) {
            A[kc][e]     = (_Float16)zrow[(size_t)(k0 + half * 16 + e)      * HW];
            A[kc][e + 8] = (_Float16)zrow[(size_t)(k0 + half * 16 + 8 + e) * HW];
        }
    }

    float best[8];
    int   bidx[8];
#pragma unroll
    for (int r = 0; r < 8; ++r) { best[r] = 3.0e38f; bidx[r] = 0; }

    // ---- Prologue: stage code tile 0 into Bbuf[0] ---------------------------
    {
        const unsigned ldsa = (unsigned)(uintptr_t)(&Bbuf[0][0]) + (unsigned)tid * 16u;
        const unsigned gof  = (unsigned)tid * 16u;
        ASYNC_CP16(ldsa, gof, cb16, 0);
        ASYNC_CP16(ldsa, gof, cb16, 2048);
        ASYNC_CP16(ldsa, gof, cb16, 4096);
        ASYNC_CP16(ldsa, gof, cb16, 6144);
        ASYNC_CP16(ldsa, gof, cb16, 8192);
        ASYNC_CP16(ldsa, gof, cb16, 10240);
        ASYNC_CP16(ldsa, gof, cb16, 12288);
        ASYNC_CP16(ldsa, gof, cb16, 14336);
        wait_async0();
        __syncthreads();
    }

    int p = 0;
    for (int cb = 0; cb < K_CODES; cb += TILE_CODES) {
        // ---- Kick off async DMA of the NEXT code tile into the other buffer -
        const int cbN = cb + TILE_CODES;
        if (cbN < K_CODES) {
            const unsigned ldsa =
                (unsigned)(uintptr_t)(&Bbuf[p ^ 1][0]) + (unsigned)tid * 16u;
            const unsigned gof = (unsigned)cbN * 512u + (unsigned)tid * 16u;
            ASYNC_CP16(ldsa, gof, cb16, 0);
            ASYNC_CP16(ldsa, gof, cb16, 2048);
            ASYNC_CP16(ldsa, gof, cb16, 4096);
            ASYNC_CP16(ldsa, gof, cb16, 6144);
            ASYNC_CP16(ldsa, gof, cb16, 8192);
            ASYNC_CP16(ldsa, gof, cb16, 10240);
            ASYNC_CP16(ldsa, gof, cb16, 12288);
            ASYNC_CP16(ldsa, gof, cb16, 14336);
        }

        // ---- Compute two 16x16 subtiles from the current LDS buffer ---------
#pragma unroll
        for (int s = 0; s < 2; ++s) {
            const int slot = s * 16 + sub;                 // lane's column slot
            const _Float16* __restrict__ bp = &Bbuf[p][slot * C_DIM];

            v8f acc = {};
#pragma unroll
            for (int kc = 0; kc < 8; ++kc) {
                const v16h B = *(const v16h*)(bp + kc * 32 + half * 16);
                acc = __builtin_amdgcn_wmma_f32_16x16x32_f16(
                        false, A[kc], false, B, (short)0, acc, false, false);
            }

            const int code = cb + slot;
            const float cn = cnorm[code];
#pragma unroll
            for (int r = 0; r < 8; ++r) {
                const float val = cn - 2.0f * acc[r];      // ||E||^2 - 2 z.E
                if (val < best[r]) { best[r] = val; bidx[r] = code; }
            }
        }

        // ---- Close the double buffer ---------------------------------------
        wait_async0();
        __syncthreads();
        p ^= 1;
    }

    // ---- Reduce across the 16 lanes holding the same rows (N dimension) ----
#pragma unroll
    for (int m = 1; m < 16; m <<= 1) {
#pragma unroll
        for (int r = 0; r < 8; ++r) {
            const float ov = __shfl_xor(best[r], m, 16);
            const int   oi = __shfl_xor(bidx[r], m, 16);
            if (ov < best[r] || (ov == best[r] && oi < bidx[r])) {
                best[r] = ov; bidx[r] = oi;
            }
        }
    }
    if (sub == 0) {
#pragma unroll
        for (int r = 0; r < 8; ++r)
            idx_out[rowBase + half * 8 + r] = bidx[r];
    }
}

// ---------------------------------------------------------------------------
// Kernel 3: gather z_q, write z_q_st ([B,C,H,W] layout), one-hot encodings,
// loss partial sums, code histogram.  grid = N_VECS blocks, 256 threads.
// ---------------------------------------------------------------------------
__global__ void vq_gather_kernel(const float* __restrict__ z,
                                 const float* __restrict__ codebook,
                                 const int* __restrict__ idx_in,
                                 int* __restrict__ counts,
                                 float* __restrict__ lossacc,
                                 float* __restrict__ out_zq,
                                 float* __restrict__ out_enc) {
    const int n = blockIdx.x;
    const int t = threadIdx.x;
    const int idx = idx_in[n];

    const int b  = n >> 10;
    const int hw = n & (HW - 1);
    const size_t off = (size_t)b * CHW + (size_t)t * HW + hw;

    const float zq = codebook[(size_t)idx * C_DIM + t];
    const float ze = z[off];
    out_zq[off] = zq;                     // straight-through output == z_q
    const float d = zq - ze;

    __shared__ float red[256];
    red[t] = d * d;
    __syncthreads();
    for (int s = 128; s > 0; s >>= 1) {
        if (t < s) red[t] += red[t + s];
        __syncthreads();
    }
    if (t == 0) {
        atomicAdd(lossacc, red[0]);
        atomicAdd(counts + idx, 1);
    }

    // one-hot row, coalesced: 8 strided sweeps of 256 floats
    float* __restrict__ enc = out_enc + (size_t)n * K_CODES;
#pragma unroll
    for (int j = 0; j < 8; ++j) {
        const int k = j * 256 + t;
        enc[k] = (k == idx) ? 1.0f : 0.0f;
    }
}

// ---------------------------------------------------------------------------
// Kernel 4: finalize loss + perplexity scalars. 1 block, 256 threads.
// ---------------------------------------------------------------------------
__global__ void vq_final_kernel(const float* __restrict__ lossacc,
                                const int* __restrict__ counts,
                                float* __restrict__ out_scalars) {
    const int t = threadIdx.x;
    float s = 0.0f;
#pragma unroll
    for (int j = 0; j < 8; ++j) {
        const float p = (float)counts[j * 256 + t] * (1.0f / (float)N_VECS);
        s += p * __logf(p + 1e-10f);
    }
    __shared__ float red[256];
    red[t] = s;
    __syncthreads();
    for (int st = 128; st > 0; st >>= 1) {
        if (t < st) red[t] += red[t + st];
        __syncthreads();
    }
    if (t == 0) {
        out_scalars[0] = lossacc[0] * (1.25f / ((float)N_VECS * (float)C_DIM));
        out_scalars[1] = __expf(-red[0]);
    }
}

// ---------------------------------------------------------------------------
extern "C" void kernel_launch(void* const* d_in, const int* in_sizes, int n_in,
                              void* d_out, int out_size, void* d_ws, size_t ws_size,
                              hipStream_t stream) {
    const float* z        = (const float*)d_in[0];   // [64,256,32,32]
    const float* codebook = (const float*)d_in[1];   // [2048,256]
    float* out = (float*)d_out;

    // workspace layout
    char* ws = (char*)d_ws;
    _Float16* cb16   = (_Float16*)(ws);                       // 1,048,576 B
    float*    cnorm  = (float*)   (ws + 1048576);             //     8,192 B
    int*      idx    = (int*)     (ws + 1056768);             //   262,144 B
    int*      counts = (int*)     (ws + 1318912);             //     8,192 B
    float*    lossw  = (float*)   (ws + 1327104);             //         4 B

    // output layout (flat, return order): z_q_st | loss | perplexity | encodings
    float* out_zq      = out;
    float* out_scalars = out + (size_t)16777216;
    float* out_enc     = out + (size_t)16777218;

    vq_prep_kernel  <<<K_CODES, 256, 0, stream>>>(codebook, cb16, cnorm, counts, lossw);
    vq_argmin_kernel<<<(N_VECS / 16) / 4, 128, 0, stream>>>(z, cb16, cnorm, idx);
    vq_gather_kernel<<<N_VECS, 256, 0, stream>>>(z, codebook, idx, counts, lossw,
                                                 out_zq, out_enc);
    vq_final_kernel <<<1, 256, 0, stream>>>(lossw, counts, out_scalars);
}
